// GNN_53704271069238
// MI455X (gfx1250) — compile-verified, matrix operand order checked by reading, not run on hardware
//
#include <hip/hip_runtime.h>
#include <hip/hip_bf16.h>

typedef __attribute__((ext_vector_type(16))) _Float16 v16h;
typedef __attribute__((ext_vector_type(2)))  _Float16 v2h;
typedef __attribute__((ext_vector_type(8)))  float    v8f;
typedef __attribute__((ext_vector_type(4)))  unsigned int uint32x4;
typedef __attribute__((ext_vector_type(8)))  int      int32x8;
typedef __attribute__((ext_vector_type(4)))  int      int32x4;

#define NN   50000
#define EE   400000
#define FIN  16
#define FEA  8
#define HH   32
#define OUTF 16
#define GG   64
#define BPACK_N (2*17*32*16)   // (c, kk, lane, 16 halfs) = 17408 halfs = 34816 B
#define MSG_BLOCKS 391         // persistent blocks (8 waves each) for the edge GEMM

// ---------------------------------------------------------------------------
// zero
__global__ void zero_kernel(float* __restrict__ p, long n) {
    long i = (long)blockIdx.x * blockDim.x + threadIdx.x;
    if (i < n) p[i] = 0.0f;
}

// ---------------------------------------------------------------------------
// Pre-pack W2' (and folded bias block kk==16) into WMMA B-fragment order.
// B is 32(K)x16(N) f16; layout mirrors A: lane<16 -> col=lane,
// K-locals {0..7,16..23}; lane>=16 -> col=lane-16, K-locals {8..15,24..31}.
__global__ void pack_b_kernel(const float* __restrict__ e_w2,   // [32,512]
                              const float* __restrict__ e_b2,   // [512]
                              _Float16* __restrict__ bpack) {
    int t = blockIdx.x * blockDim.x + threadIdx.x;
    if (t >= BPACK_N) return;
    int j    = t & 15;
    int lane = (t >> 4) & 31;
    int rest = t >> 9;
    int kk   = rest % 17;
    int c    = rest / 17;
    int loc  = (lane < 16) ? ((j < 8) ? j : 16 + (j - 8))
                           : ((j < 8) ? 8 + j : 24 + (j - 8));
    int h = c * 16 + (lane & 15);
    float v;
    if (kk < 16) {
        int k  = kk * 32 + loc;       // global K index into W2' (K = kq*16 + f)
        int kq = k >> 4;
        int f  = k & 15;
        v = e_w2[kq * 512 + f * 32 + h];
    } else {
        v = (loc < 16) ? e_b2[loc * 32 + h] : 0.0f;   // bias-fold block
    }
    bpack[t] = (_Float16)v;
}

// ---------------------------------------------------------------------------
// Edge in-degree counts (== NNConv cnt, and GCN deg-1)
__global__ void edge_count_kernel(const int* __restrict__ ei, float* __restrict__ cnt) {
    int e = blockIdx.x * blockDim.x + threadIdx.x;
    if (e < EE) atomicAdd(cnt + ei[EE + e], 1.0f);
}

// ---------------------------------------------------------------------------
// THE hot kernel: fused edge-MLP + per-edge message as one big f16 WMMA GEMM.
// Persistent: each of 8 waves loops over 16-edge tiles; packed B fragments
// staged into LDS once per block via the Tensor Data Mover.
// Register plan (cap 256/lane via launch_bounds):
//   - W1 weights (f16 pairs) hoisted from LDS into ~128 VGPRs (wanted),
//   - B fragments forced to stay as in-loop ds_load_b128 via opaque offset.
__global__ __launch_bounds__(256, 4) void edge_msg_kernel(
        const float* __restrict__ x,          // [N,16]
        const float* __restrict__ edge_attr,  // [E,8]
        const int*   __restrict__ ei,         // [2,E]
        const float* __restrict__ e_w1,       // [8,32]
        const float* __restrict__ e_b1,       // [32]
        const _Float16* __restrict__ bpack,   // [BPACK_N]
        float* __restrict__ agg)              // [N,32] (atomic accumulate)
{
    __shared__ __align__(16) _Float16 sB[BPACK_N];  // 34,816 B packed B frags
    __shared__ __align__(16) _Float16 sW1h[8 * 32];
    __shared__ __align__(16) _Float16 sB1h[32];

    const int lane = threadIdx.x & 31;
    const int wave = threadIdx.x >> 5;

    // --- TDM: DMA bpack -> LDS (1-D tile, 4352 x 8B), issued by wave 0 only
    if (wave == 0) {
        unsigned lds_off = (unsigned)(size_t)(&sB[0]);
        unsigned long long ga = (unsigned long long)(size_t)bpack;
        uint32x4 g0 = { 1u,                                   // count=1, user D#
                        lds_off,                              // lds_addr
                        (unsigned)(ga & 0xffffffffu),         // global_addr lo
                        (unsigned)((ga >> 32) & 0x01ffffffu) | (2u << 30) }; // hi + type=2
        int32x8 g1 = { (int)(3u << 16),        // data_size=3 (8B), mask=0
                       (int)(4352u << 16),     // tensor_dim0 lo16 in [63:48]
                       (int)(1u << 16),        // tensor_dim1 = 1
                       (int)(4352u << 16),     // tile_dim0 in [127:112]
                       1,                      // tile_dim1 = 1
                       4352,                   // tensor_dim0_stride lo32
                       0, 0 };
        int32x4 gz4 = { 0, 0, 0, 0 };
        int32x8 gz8 = { 0, 0, 0, 0, 0, 0, 0, 0 };
        __builtin_amdgcn_tensor_load_to_lds(g0, g1, gz4, gz4, gz8, 0);
        __builtin_amdgcn_s_wait_tensorcnt(0);
    }
    for (int t = threadIdx.x; t < 256; t += blockDim.x) sW1h[t] = (_Float16)e_w1[t];
    if (threadIdx.x < 32) sB1h[threadIdx.x] = (_Float16)e_b1[threadIdx.x];
    __syncthreads();

    const int nTiles = EE / 16;            // 25000
    const int stride = gridDim.x * 8;
    const int row    = lane & 15;
    const v16h* bp   = (const v16h*)sB;
    unsigned bzero   = 0;

    #pragma unroll 1
    for (int tile = blockIdx.x * 8 + wave; tile < nTiles; tile += stride) {
        const int eb = tile * 16;
        const int e  = eb + row;           // E divisible by 16

        // opaque zero: stops LICM from hoisting (and then spilling) B frags
        unsigned bofs = bzero;
        asm volatile("" : "+v"(bofs));
        const v16h* bpt = bp + bofs;

        // speculative prefetch of our next tile's edge_attr (global_prefetch_b8)
        {
            int en = (tile + stride) * 16 + row;
            if (en < EE) __builtin_prefetch(edge_attr + (size_t)en * 8, 0, 0);
        }

        // --- layer-1 of edge MLP in packed f16: h1 = relu(ea @ W1 + b1)
        _Float16 eah[8];
        {
            const float4* p = (const float4*)(edge_attr + (size_t)e * 8);
            float4 a0 = p[0], a1 = p[1];
            eah[0]=(_Float16)a0.x; eah[1]=(_Float16)a0.y; eah[2]=(_Float16)a0.z; eah[3]=(_Float16)a0.w;
            eah[4]=(_Float16)a1.x; eah[5]=(_Float16)a1.y; eah[6]=(_Float16)a1.z; eah[7]=(_Float16)a1.w;
        }
        v2h h1p[16];                       // pairs (h1[2k], h1[2k+1])
        #pragma unroll
        for (int jp = 0; jp < 16; ++jp) {
            v2h acc = *(const v2h*)&sB1h[2 * jp];
            #pragma unroll
            for (int i = 0; i < 8; ++i) {
                v2h w = *(const v2h*)&sW1h[i * 32 + 2 * jp];
                v2h eb2 = { eah[i], eah[i] };
                acc = acc + eb2 * w;       // v_pk_fma_f16
            }
            _Float16 z = (_Float16)0.0f;
            v2h r;
            r[0] = (acc[0] > z) ? acc[0] : z;
            r[1] = (acc[1] > z) ? acc[1] : z;
            h1p[jp] = r;
        }

        // --- gathered x half-row (lane<16 -> f 0..7, else f 8..15)
        const int src = ei[e];
        _Float16 xh[8];
        {
            const float* xr = x + (size_t)src * 16 + ((lane >= 16) ? 8 : 0);
            float4 x0 = *(const float4*)(xr);
            float4 x1 = *(const float4*)(xr + 4);
            xh[0]=(_Float16)x0.x; xh[1]=(_Float16)x0.y; xh[2]=(_Float16)x0.z; xh[3]=(_Float16)x0.w;
            xh[4]=(_Float16)x1.x; xh[5]=(_Float16)x1.y; xh[6]=(_Float16)x1.z; xh[7]=(_Float16)x1.w;
        }

        // --- GEMM: msg[16,32] = P[16,512+pad] @ W2'[512+pad,32]
        v8f acc0 = {};
        v8f acc1 = {};
        #pragma unroll
        for (int kk = 0; kk < 17; ++kk) {
            v16h a;
            if (kk < 16) {
                _Float16 lo = h1p[kk][0];
                _Float16 hi = h1p[kk][1];
                #pragma unroll
                for (int j = 0; j < 8; ++j) { a[j] = lo * xh[j]; a[8 + j] = hi * xh[j]; }
            } else {        // bias-fold block: A2 = [x | 0]
                #pragma unroll
                for (int j = 0; j < 8; ++j) { a[j] = xh[j]; a[8 + j] = (_Float16)0.0f; }
            }
            v16h b0 = bpt[(0 * 17 + kk) * 32 + lane];
            v16h b1 = bpt[(1 * 17 + kk) * 32 + lane];
            acc0 = __builtin_amdgcn_wmma_f32_16x16x32_f16(false, a, false, b0, (short)0, acc0, false, false);
            acc1 = __builtin_amdgcn_wmma_f32_16x16x32_f16(false, a, false, b1, (short)0, acc1, false, false);
        }

        // --- scatter-add D fragments: lane<16 holds rows 0..7, col=lane
        const int col   = lane & 15;
        const int mbase = (lane < 16) ? 0 : 8;
        #pragma unroll
        for (int r = 0; r < 8; ++r) {
            int ee2 = eb + mbase + r;
            int dst = ei[EE + ee2];
            atomicAdd(agg + (size_t)dst * 32 + col,      acc0[r]);
            atomicAdd(agg + (size_t)dst * 32 + 16 + col, acc1[r]);
        }
    }
}

// ---------------------------------------------------------------------------
// h0 = relu(x @ nn_root + agg/max(cnt,1) + nn_bias); accumulate BN1 stats
__global__ __launch_bounds__(256) void nnconv_finalize_kernel(
        const float* __restrict__ x, const float* __restrict__ agg,
        const float* __restrict__ cnt,
        const float* __restrict__ nn_root, const float* __restrict__ nn_bias,
        float* __restrict__ h0, float* __restrict__ stats)
{
    __shared__ float sR[16 * 32];
    __shared__ float sBias[32];
    __shared__ float sSum[32], sSq[32];
    for (int t = threadIdx.x; t < 512; t += blockDim.x) sR[t] = nn_root[t];
    if (threadIdx.x < 32) { sBias[threadIdx.x] = nn_bias[threadIdx.x];
                            sSum[threadIdx.x] = 0.0f; sSq[threadIdx.x] = 0.0f; }
    __syncthreads();
    int n = blockIdx.x * blockDim.x + threadIdx.x;
    if (n < NN) {
        float xr[16];
        const float4* xp = (const float4*)(x + (size_t)n * 16);
        #pragma unroll
        for (int q = 0; q < 4; ++q) {
            float4 v = xp[q];
            xr[q*4+0]=v.x; xr[q*4+1]=v.y; xr[q*4+2]=v.z; xr[q*4+3]=v.w;
        }
        float inv = 1.0f / fmaxf(cnt[n], 1.0f);
        const float* ap = agg + (size_t)n * 32;
        float*       op = h0  + (size_t)n * 32;
        #pragma unroll
        for (int j = 0; j < 32; ++j) {
            float s = sBias[j] + ap[j] * inv;
            #pragma unroll
            for (int i = 0; i < 16; ++i) s += xr[i] * sR[i * 32 + j];
            s = fmaxf(s, 0.0f);
            op[j] = s;
            atomicAdd(&sSum[j], s);
            atomicAdd(&sSq[j], s * s);
        }
    }
    __syncthreads();
    if (threadIdx.x < 32) {
        atomicAdd(stats + threadIdx.x,      sSum[threadIdx.x]);
        atomicAdd(stats + 32 + threadIdx.x, sSq[threadIdx.x]);
    }
}

// ---------------------------------------------------------------------------
// hw = BN(hin) @ w  via WMMA (K=32, one 16x16x32 per column tile).
__global__ __launch_bounds__(128) void bn_matmul_kernel(
        const float* __restrict__ hin,   // [N,32]
        const float* __restrict__ stats, // [64] sum|sumsq
        const float* __restrict__ gamma, const float* __restrict__ beta,
        const float* __restrict__ w,     // [32,32]
        float* __restrict__ hw)          // [N,32]
{
    __shared__ _Float16 sB[2 * 32 * 16];
    __shared__ float sScale[32], sShift[32];
    if (threadIdx.x < 32) {
        int ch = threadIdx.x;
        float mean = stats[ch] * (1.0f / NN);
        float var  = stats[32 + ch] * (1.0f / NN) - mean * mean;
        float sc   = rsqrtf(var + 1e-5f) * gamma[ch];
        sScale[ch] = sc;
        sShift[ch] = beta[ch] - mean * sc;
    }
    for (int t = threadIdx.x; t < 2 * 32 * 16; t += blockDim.x) {
        int j = t & 15, lane = (t >> 4) & 31, c = t >> 9;
        int loc = (lane < 16) ? ((j < 8) ? j : 16 + (j - 8))
                              : ((j < 8) ? 8 + j : 24 + (j - 8));
        sB[t] = (_Float16)w[loc * 32 + c * 16 + (lane & 15)];
    }
    __syncthreads();

    const int lane = threadIdx.x & 31;
    const int wave = threadIdx.x >> 5;
    const int tile = blockIdx.x * 4 + wave;
    const int nb   = tile * 16;
    if (nb >= NN) return;

    const int rowe = nb + (lane & 15);          // N divisible by 16
    const int off  = (lane >= 16) ? 8 : 0;
    const float* hp = hin + (size_t)rowe * 32;
    v16h a;
    #pragma unroll
    for (int j = 0; j < 8; ++j) {
        int ch = off + j;
        a[j] = (_Float16)(hp[ch] * sScale[ch] + sShift[ch]);
    }
    #pragma unroll
    for (int j = 0; j < 8; ++j) {
        int ch = 16 + off + j;
        a[8 + j] = (_Float16)(hp[ch] * sScale[ch] + sShift[ch]);
    }
    v8f acc0 = {};
    v8f acc1 = {};
    const v16h* bp = (const v16h*)sB;
    acc0 = __builtin_amdgcn_wmma_f32_16x16x32_f16(false, a, false, bp[lane],      (short)0, acc0, false, false);
    acc1 = __builtin_amdgcn_wmma_f32_16x16x32_f16(false, a, false, bp[32 + lane], (short)0, acc1, false, false);

    const int col   = lane & 15;
    const int mbase = (lane < 16) ? 0 : 8;
    #pragma unroll
    for (int r = 0; r < 8; ++r) {
        size_t node = (size_t)(nb + mbase + r);
        hw[node * 32 + col]      = acc0[r];
        hw[node * 32 + 16 + col] = acc1[r];
    }
}

// ---------------------------------------------------------------------------
// GCN edge aggregation: gagg[dst] += hw[src] * rsqrt(deg_s)*rsqrt(deg_d)
__global__ __launch_bounds__(256) void gcn_edge_agg_kernel(
        const int* __restrict__ ei, const float* __restrict__ cnt,
        const float* __restrict__ hw, float* __restrict__ gagg)
{
    int e = blockIdx.x * blockDim.x + threadIdx.x;
    if (e >= EE) return;
    int s = ei[e], d = ei[EE + e];
    float norm = rsqrtf(cnt[s] + 1.0f) * rsqrtf(cnt[d] + 1.0f);
    const float4* hp = (const float4*)(hw + (size_t)s * 32);
    float* gp = gagg + (size_t)d * 32;
    #pragma unroll
    for (int q = 0; q < 8; ++q) {
        float4 v = hp[q];
        atomicAdd(gp + q * 4 + 0, v.x * norm);
        atomicAdd(gp + q * 4 + 1, v.y * norm);
        atomicAdd(gp + q * 4 + 2, v.z * norm);
        atomicAdd(gp + q * 4 + 3, v.w * norm);
    }
}

// ---------------------------------------------------------------------------
// g = relu(gagg + hw/deg + bias); accumulate BN stats
__global__ __launch_bounds__(256) void gcn_finalize_kernel(
        const float* __restrict__ gagg, const float* __restrict__ hw,
        const float* __restrict__ cnt,  const float* __restrict__ bias,
        float* __restrict__ gout, float* __restrict__ stats)
{
    __shared__ float sBias[32];
    __shared__ float sSum[32], sSq[32];
    if (threadIdx.x < 32) { sBias[threadIdx.x] = bias[threadIdx.x];
                            sSum[threadIdx.x] = 0.0f; sSq[threadIdx.x] = 0.0f; }
    __syncthreads();
    int n = blockIdx.x * blockDim.x + threadIdx.x;
    if (n < NN) {
        float invdeg = 1.0f / (cnt[n] + 1.0f);
        const float* ga = gagg + (size_t)n * 32;
        const float* hp = hw   + (size_t)n * 32;
        float*       op = gout + (size_t)n * 32;
        #pragma unroll
        for (int j = 0; j < 32; ++j) {
            float v = fmaxf(ga[j] + hp[j] * invdeg + sBias[j], 0.0f);
            op[j] = v;
            atomicAdd(&sSum[j], v);
            atomicAdd(&sSq[j], v * v);
        }
    }
    __syncthreads();
    if (threadIdx.x < 32) {
        atomicAdd(stats + threadIdx.x,      sSum[threadIdx.x]);
        atomicAdd(stats + 32 + threadIdx.x, sSq[threadIdx.x]);
    }
}

// ---------------------------------------------------------------------------
// apply BN3 and pool per graph (sum + counts)
__global__ __launch_bounds__(256) void pool_kernel(
        const float* __restrict__ g2, const float* __restrict__ stats,
        const float* __restrict__ gamma, const float* __restrict__ beta,
        const int* __restrict__ batch,
        float* __restrict__ pool, float* __restrict__ gcnt)
{
    __shared__ float sScale[32], sShift[32];
    if (threadIdx.x < 32) {
        int ch = threadIdx.x;
        float mean = stats[ch] * (1.0f / NN);
        float var  = stats[32 + ch] * (1.0f / NN) - mean * mean;
        float sc   = rsqrtf(var + 1e-5f) * gamma[ch];
        sScale[ch] = sc;
        sShift[ch] = beta[ch] - mean * sc;
    }
    __syncthreads();
    int n = blockIdx.x * blockDim.x + threadIdx.x;
    if (n >= NN) return;
    int b = batch[n];
    atomicAdd(gcnt + b, 1.0f);
    const float* gp = g2 + (size_t)n * 32;
    float* pp = pool + (size_t)b * 32;
    #pragma unroll
    for (int j = 0; j < 32; ++j)
        atomicAdd(pp + j, gp[j] * sScale[j] + sShift[j]);
}

// ---------------------------------------------------------------------------
// MLP head over G=64 graphs (tiny)
__global__ void head_kernel(const float* __restrict__ pool, const float* __restrict__ gcnt,
                            const float* __restrict__ fc1w, const float* __restrict__ fc1b,
                            const float* __restrict__ fc2w, const float* __restrict__ fc2b,
                            float* __restrict__ out)
{
    int g = threadIdx.x;
    if (g >= GG) return;
    float inv = 1.0f / fmaxf(gcnt[g], 1.0f);
    float p[32], t[32];
    #pragma unroll
    for (int j = 0; j < 32; ++j) p[j] = pool[g * 32 + j] * inv;
    #pragma unroll
    for (int k = 0; k < 32; ++k) {
        float s = fc1b[k];
        #pragma unroll
        for (int j = 0; j < 32; ++j) s += p[j] * fc1w[j * 32 + k];
        t[k] = fmaxf(s, 0.0f);
    }
    #pragma unroll
    for (int o = 0; o < 16; ++o) {
        float s = fc2b[o];
        #pragma unroll
        for (int k = 0; k < 32; ++k) s += t[k] * fc2w[k * 16 + o];
        out[g * 16 + o] = s;
    }
}

// ---------------------------------------------------------------------------
extern "C" void kernel_launch(void* const* d_in, const int* in_sizes, int n_in,
                              void* d_out, int out_size, void* d_ws, size_t ws_size,
                              hipStream_t stream) {
    const float* x         = (const float*)d_in[0];
    const float* edge_attr = (const float*)d_in[1];
    const int*   ei        = (const int*)  d_in[2];
    const int*   batch     = (const int*)  d_in[3];
    const float* e_w1      = (const float*)d_in[4];
    const float* e_b1      = (const float*)d_in[5];
    const float* e_w2      = (const float*)d_in[6];
    const float* e_b2      = (const float*)d_in[7];
    const float* nn_root   = (const float*)d_in[8];
    const float* nn_bias   = (const float*)d_in[9];
    const float* g1w  = (const float*)d_in[10];
    const float* g1b  = (const float*)d_in[11];
    const float* g2w  = (const float*)d_in[12];
    const float* g2b  = (const float*)d_in[13];
    const float* bn1g = (const float*)d_in[14];
    const float* bn1b = (const float*)d_in[15];
    const float* bn2g = (const float*)d_in[16];
    const float* bn2b = (const float*)d_in[17];
    const float* bn3g = (const float*)d_in[18];
    const float* bn3b = (const float*)d_in[19];
    const float* fc1w = (const float*)d_in[20];
    const float* fc1b = (const float*)d_in[21];
    const float* fc2w = (const float*)d_in[22];
    const float* fc2b = (const float*)d_in[23];
    float* out = (float*)d_out;

    // ---- workspace carve-out (~26 MB) ----
    char* base = (char*)d_ws;
    size_t off = 0;
    auto take = [&](size_t bytes) -> void* {
        void* p = base + off;
        off = (off + bytes + 255) & ~(size_t)255;
        return p;
    };
    float*    cnt   = (float*)take((size_t)NN * 4);
    float*    st1   = (float*)take(64 * 4);
    float*    st2   = (float*)take(64 * 4);
    float*    st3   = (float*)take(64 * 4);
    float*    pool  = (float*)take((size_t)GG * HH * 4);
    float*    gcnt  = (float*)take(GG * 4);
    _Float16* bpack = (_Float16*)take((size_t)BPACK_N * 2);
    float*    W0    = (float*)take((size_t)NN * HH * 4);
    float*    W1    = (float*)take((size_t)NN * HH * 4);
    float*    W2    = (float*)take((size_t)NN * HH * 4);
    float*    W3    = (float*)take((size_t)NN * HH * 4);

    auto z = [&](float* p, long n) {
        zero_kernel<<<(unsigned)((n + 255) / 256), 256, 0, stream>>>(p, n);
    };
    z(cnt, NN); z(st1, 64); z(st2, 64); z(st3, 64);
    z(pool, GG * HH); z(gcnt, GG);
    z(W0, (long)NN * HH);          // agg
    z(W3, (long)NN * HH);          // gagg1

    pack_b_kernel<<<(BPACK_N + 255) / 256, 256, 0, stream>>>(e_w2, e_b2, bpack);
    edge_count_kernel<<<(EE + 255) / 256, 256, 0, stream>>>(ei, cnt);

    // fused NNConv messages (persistent WMMA GEMM), atomically into W0
    edge_msg_kernel<<<MSG_BLOCKS, 256, 0, stream>>>(x, edge_attr, ei, e_w1, e_b1, bpack, W0);

    // h0 = relu(x@root + agg/cnt + b) -> W1, BN1 stats
    nnconv_finalize_kernel<<<(NN + 255) / 256, 256, 0, stream>>>(x, W0, cnt, nn_root, nn_bias, W1, st1);

    // layer 1: hw1 = BN1(h0)@W -> W2 ; gagg1 -> W3 ; g1 -> W0 (+BN2 stats)
    bn_matmul_kernel<<<(NN / 16 + 3) / 4, 128, 0, stream>>>(W1, st1, bn1g, bn1b, g1w, W2);
    gcn_edge_agg_kernel<<<(EE + 255) / 256, 256, 0, stream>>>(ei, cnt, W2, W3);
    gcn_finalize_kernel<<<(NN + 255) / 256, 256, 0, stream>>>(W3, W2, cnt, g1b, W0, st2);

    // layer 2: hw2 = BN2(g1)@W -> W1 ; gagg2 -> W2 (zeroed) ; g2 -> W3 (+BN3 stats)
    bn_matmul_kernel<<<(NN / 16 + 3) / 4, 128, 0, stream>>>(W0, st2, bn2g, bn2b, g2w, W1);
    z(W2, (long)NN * HH);
    gcn_edge_agg_kernel<<<(EE + 255) / 256, 256, 0, stream>>>(ei, cnt, W1, W2);
    gcn_finalize_kernel<<<(NN + 255) / 256, 256, 0, stream>>>(W2, W1, cnt, g2b, W3, st3);

    // BN3 + global mean pool + head
    pool_kernel<<<(NN + 255) / 256, 256, 0, stream>>>(W3, st3, bn3g, bn3b, batch, pool, gcnt);
    head_kernel<<<1, 64, 0, stream>>>(pool, gcnt, fc1w, fc1b, fc2w, fc2b, out);
}
